// MyDense_79285096284485
// MI455X (gfx1250) — 8-bit matrix operand order repaired and checked by reading, compiled, not run on hardware
//
#include <hip/hip_runtime.h>
#include <hip/hip_bf16.h>

typedef __attribute__((ext_vector_type(16))) int   v16i;
typedef __attribute__((ext_vector_type(8)))  float v8f;
typedef unsigned long long u64;

constexpr int Mdim = 8192;
constexpr int Kdim = 4096;
constexpr int Ndim = 4096;
constexpr float E4M3_MAX = 448.0f;

__device__ __forceinline__ unsigned int f32_to_e4m3(float x) {
    x = fminf(fmaxf(x, -E4M3_MAX), E4M3_MAX);
    unsigned u = __float_as_uint(x);
    unsigned s = (u >> 24) & 0x80u;
    unsigned au = u & 0x7FFFFFFFu;
    if (au < 0x3C800000u) {
        float a = __uint_as_float(au);
        unsigned m = (unsigned)rintf(a * 512.0f);
        return s | m;
    }
    unsigned r = au + 0x0007FFFFu + ((au >> 20) & 1u);
    int e4 = ((int)(r >> 23) - 127) + 7;
    unsigned mant = (r >> 20) & 7u;
    if (e4 > 15) { e4 = 15; mant = 6; }
    return s | ((unsigned)e4 << 3) | mant;
}

__device__ __forceinline__ unsigned int pack4(float a, float b, float c, float d, float sc) {
    return f32_to_e4m3(a * sc)        |
          (f32_to_e4m3(b * sc) << 8)  |
          (f32_to_e4m3(c * sc) << 16) |
          (f32_to_e4m3(d * sc) << 24);
}

__global__ void scale_update(const float* __restrict__ amax_hist,
                             const float* __restrict__ scale_in,
                             float* __restrict__ hdr) {
    if (threadIdx.x == 0) {
        float sf01[2];
        for (int i = 0; i < 2; ++i) {
            float amax = amax_hist[i];
            float sc   = scale_in[i];
            float e    = floorf(log2f(E4M3_MAX / amax));
            float sf   = rintf(exp2f(fabsf(e)));
            if (!(amax > 0.0f)) sf = sc;
            if (!isfinite(amax)) sf = sc;
            if (e < 0.0f) sf = 1.0f / sf;
            sf01[i] = sf;
            hdr[i] = sf;
        }
        hdr[2] = 1.0f / (sf01[0] * sf01[1]);
    }
}

__global__ void __launch_bounds__(256) quant_x(const float* __restrict__ x,
                                               const float* __restrict__ hdr,
                                               unsigned char* __restrict__ qx) {
    const float sc = hdr[0];
    long long t = (long long)blockIdx.x * 256 + threadIdx.x;
    const float4* xv = (const float4*)x;
    float4 a = xv[4 * t + 0];
    float4 b = xv[4 * t + 1];
    float4 c = xv[4 * t + 2];
    float4 d = xv[4 * t + 3];
    uint4 o;
    o.x = pack4(a.x, a.y, a.z, a.w, sc);
    o.y = pack4(b.x, b.y, b.z, b.w, sc);
    o.z = pack4(c.x, c.y, c.z, c.w, sc);
    o.w = pack4(d.x, d.y, d.z, d.w, sc);
    ((uint4*)qx)[t] = o;
}

__global__ void __launch_bounds__(256) quant_wT(const float* __restrict__ w,
                                                const float* __restrict__ hdr,
                                                unsigned char* __restrict__ qwT) {
    const float sc = hdr[1];
    long long t = (long long)blockIdx.x * 256 + threadIdx.x;
    int k4 = (int)(t & (Kdim / 4 - 1));
    int n  = (int)(t >> 10);
    int k  = k4 * 4;
    float v0 = w[(long long)(k + 0) * Ndim + n];
    float v1 = w[(long long)(k + 1) * Ndim + n];
    float v2 = w[(long long)(k + 2) * Ndim + n];
    float v3 = w[(long long)(k + 3) * Ndim + n];
    *(unsigned int*)(qwT + (long long)n * Kdim + k) = pack4(v0, v1, v2, v3, sc);
}

__device__ __forceinline__ void stage_tiles_async(const unsigned char* __restrict__ gA,
                                                  const unsigned char* __restrict__ gB,
                                                  unsigned char (*As)[128],
                                                  unsigned char (*Bs)[128],
                                                  int tid) {
#pragma unroll
    for (int p = 0; p < 4; ++p) {
        int c    = tid + p * 256;
        int row  = c >> 3;
        int colb = (c & 7) << 4;
        const unsigned char* ga = gA + (long long)row * Kdim + colb;
        const unsigned char* gb = gB + (long long)row * Kdim + colb;
        unsigned la = (unsigned)(size_t)&As[row][colb];
        unsigned lb = (unsigned)(size_t)&Bs[row][colb];
        asm volatile("global_load_async_to_lds_b128 %0, %1, off"
                     :: "v"(la), "v"(ga) : "memory");
        asm volatile("global_load_async_to_lds_b128 %0, %1, off"
                     :: "v"(lb), "v"(gb) : "memory");
    }
}

__global__ void __launch_bounds__(256) fp8_gemm(const unsigned char* __restrict__ qx,
                                                const unsigned char* __restrict__ qwT,
                                                const float* __restrict__ hdr,
                                                float* __restrict__ out) {
    constexpr int BM = 128, BN = 128, BK = 128;
    constexpr int NK = Kdim / BK;
    __shared__ __align__(16) unsigned char Als[2][BM][BK];
    __shared__ __align__(16) unsigned char Bls[2][BN][BK];

    const int tid  = threadIdx.x;
    const int lane = tid & 31;
    const int wave = tid >> 5;
    const int wm   = (wave & 3) * 32;
    const int wn   = (wave >> 2) * 64;
    const int m0   = blockIdx.y * BM;
    const int n0   = blockIdx.x * BN;
    const int half = lane >> 4;
    const int l16  = lane & 15;

    const float sinv = hdr[2];

    v8f acc[2][4];
#pragma unroll
    for (int r = 0; r < 2; ++r)
#pragma unroll
        for (int c = 0; c < 4; ++c)
            acc[r][c] = (v8f){0.f, 0.f, 0.f, 0.f, 0.f, 0.f, 0.f, 0.f};

    const unsigned char* gA = qx  + (long long)m0 * Kdim;
    const unsigned char* gB = qwT + (long long)n0 * Kdim;

    stage_tiles_async(gA, gB, Als[0], Bls[0], tid);

    for (int i = 0; i < NK; ++i) {
        const int cur = i & 1;
        if (i + 1 < NK) {
            stage_tiles_async(gA + (i + 1) * BK, gB + (i + 1) * BK,
                              Als[cur ^ 1], Bls[cur ^ 1], tid);
            asm volatile("s_wait_asynccnt 8" ::: "memory");
        } else {
            asm volatile("s_wait_asynccnt 0" ::: "memory");
        }
        __syncthreads();

        union { v16i v; u64 d[8]; } af[2];
#pragma unroll
        for (int r = 0; r < 2; ++r) {
            int row = wm + r * 16 + l16;
#pragma unroll
            for (int h = 0; h < 2; ++h)
#pragma unroll
                for (int k = 0; k < 4; ++k)
                    af[r].d[h * 4 + k] =
                        *(const u64*)(&Als[cur][row][h * 64 + (k >> 1) * 32 + half * 16 + (k & 1) * 8]);
        }

#pragma unroll
        for (int c = 0; c < 4; ++c) {
            union { v16i v; uint4 q[4]; } bf;
            int n = wn + c * 16 + l16;
#pragma unroll
            for (int j = 0; j < 4; ++j)
                bf.q[j] = *(const uint4*)(&Bls[cur][n][j * 32 + half * 16]);
#pragma unroll
            for (int r = 0; r < 2; ++r)
                acc[r][c] = __builtin_amdgcn_wmma_f32_16x16x128_fp8_fp8(
                    af[r].v, bf.v, (short)0, acc[r][c],
                     false,  false);
        }
        __syncthreads();
    }

#pragma unroll
    for (int r = 0; r < 2; ++r)
#pragma unroll
        for (int c = 0; c < 4; ++c) {
            int col = n0 + wn + c * 16 + l16;
#pragma unroll
            for (int j = 0; j < 8; ++j) {
                int row = m0 + wm + r * 16 + half * 8 + j;
                out[(long long)row * Ndim + col] = acc[r][c][j] * sinv;
            }
        }
}

extern "C" void kernel_launch(void* const* d_in, const int* in_sizes, int n_in,
                              void* d_out, int out_size, void* d_ws, size_t ws_size,
                              hipStream_t stream) {
    const float* x         = (const float*)d_in[0];
    const float* w         = (const float*)d_in[1];
    const float* amax_hist = (const float*)d_in[2];
    const float* scale     = (const float*)d_in[3];
    float* out = (float*)d_out;

    char* ws = (char*)d_ws;
    float* hdr = (float*)ws;
    unsigned char* qx  = (unsigned char*)(ws + 256);
    unsigned char* qwT = qx + (size_t)Mdim * Kdim;

    scale_update<<<1, 32, 0, stream>>>(amax_hist, scale, hdr);

    const long long xChunks = (long long)Mdim * Kdim / 16;
    quant_x<<<(unsigned)(xChunks / 256), 256, 0, stream>>>(x, hdr, qx);

    const long long wThreads = (long long)Ndim * (Kdim / 4);
    quant_wT<<<(unsigned)(wThreads / 256), 256, 0, stream>>>(w, hdr, qwT);

    dim3 grid(Ndim / 128, Mdim / 128);
    fp8_gemm<<<grid, 256, 0, stream>>>(qx, qwT, hdr, out);
}
